// TransformerDecoder_55336358641735
// MI455X (gfx1250) — matrix operand order repaired and checked by reading, not compiled, not run on hardware
//
#include <hip/hip_runtime.h>

// ---------------------------------------------------------------------------
// Types for CDNA5 WMMA (wave32): v_wmma_f32_16x16x32_bf16
// ---------------------------------------------------------------------------
typedef __attribute__((ext_vector_type(16))) __bf16 v16bf;
typedef __attribute__((ext_vector_type(8)))  __bf16 bf16x8;
typedef __attribute__((ext_vector_type(8)))  float  v8f;
typedef __attribute__((ext_vector_type(4)))  unsigned int v4u;
typedef __attribute__((ext_vector_type(8)))  int v8i;
typedef __attribute__((ext_vector_type(4)))  int v4i;

#if __has_builtin(__builtin_amdgcn_tensor_load_to_lds) && \
    __has_builtin(__builtin_amdgcn_s_wait_tensorcnt)
#define USE_TDM 1
#else
#define USE_TDM 0
#endif

// Model constants
#define LAY 6
#define DM  768
#define NH  12
#define DH  64
#define DFF 3072
#define BB  8
#define TT  512
#define SS  512
#define MM  (BB * TT)   // 4096 rows for every GEMM

__device__ __forceinline__ float gelu_f(float x) {
    const float c = 0.7978845608028654f; // sqrt(2/pi)
    return 0.5f * x * (1.0f + tanhf(c * (x + 0.044715f * x * x * x)));
}

// Butterfly max across each 16-lane row: v_permlane16_b32 (VALU, no LDS path)
__device__ __forceinline__ float rowmax16(float v) {
#if __has_builtin(__builtin_amdgcn_permlane16)
    unsigned iv, t;
    iv = __float_as_uint(v);
    t  = __builtin_amdgcn_permlane16(iv, iv, 0x67452301u, 0xEFCDAB89u, false, false); // xor 1
    v  = fmaxf(v, __uint_as_float(t));
    iv = __float_as_uint(v);
    t  = __builtin_amdgcn_permlane16(iv, iv, 0x45670123u, 0xCDEF89ABu, false, false); // xor 2
    v  = fmaxf(v, __uint_as_float(t));
    iv = __float_as_uint(v);
    t  = __builtin_amdgcn_permlane16(iv, iv, 0x01234567u, 0x89ABCDEFu, false, false); // xor 4
    v  = fmaxf(v, __uint_as_float(t));
    iv = __float_as_uint(v);
    t  = __builtin_amdgcn_permlane16(iv, iv, 0xFEDCBA98u, 0x76543210u, false, false); // xor 8
    v  = fmaxf(v, __uint_as_float(t));
    return v;
#else
    for (int off = 1; off < 16; off <<= 1) v = fmaxf(v, __shfl_xor(v, off, 32));
    return v;
#endif
}

#if USE_TDM
// Issue a TDM 2D tile load: rows x cols bf16 tile of a row-major tensor
// (row stride = strideElems) -> contiguous LDS at ldsAddr. D# per ISA 8.3/8.4.
// 6-arg builtin form: (g0, g1, g2, g3, g4, cpol).
__device__ __forceinline__ void tdm_load_tile_bf16(unsigned ldsAddr, const void* gaddr,
                                                   unsigned cols, unsigned rows,
                                                   unsigned tensorDim0,
                                                   unsigned strideElems) {
    unsigned long long ga = (unsigned long long)(size_t)gaddr;
    v4u g0;
    g0.x = 1u;                                   // count=1 (valid user descriptor)
    g0.y = ldsAddr;                              // LDS byte address
    g0.z = (unsigned)ga;                         // global_addr[31:0]
    g0.w = (unsigned)(ga >> 32) | (2u << 30);    // global_addr[56:32] | type=2 (image)
    v8i g1;
    g1[0] = (int)(1u << 16);                     // data_size=1 -> 2 bytes/elem
    g1[1] = (int)((tensorDim0 & 0xFFFFu) << 16); // tensor_dim0[15:0]
    g1[2] = (int)((tensorDim0 >> 16) | (rows << 16));       // td0[31:16] | td1[15:0]
    g1[3] = (int)((cols & 0xFFFFu) << 16);       // td1[31:16]=0 | tile_dim0
    g1[4] = (int)(rows & 0xFFFFu);               // tile_dim1 (tile_dim2=0)
    g1[5] = (int)strideElems;                    // tensor_dim0_stride[31:0]
    g1[6] = 0;                                   // stride[47:32] | td1_stride lo
    g1[7] = 0;
    v4i z4 = {0, 0, 0, 0};
    v8i z8 = {0, 0, 0, 0, 0, 0, 0, 0};
    __builtin_amdgcn_tensor_load_to_lds(g0, g1, z4, z4, z8, 0);
}
#endif

// ---------------------------------------------------------------------------
// f32 -> bf16 bulk convert (weights, memory bank)
// ---------------------------------------------------------------------------
__global__ void cvt_f32_bf16(const float* __restrict__ in, __bf16* __restrict__ out, size_t n) {
    size_t i = (size_t)blockIdx.x * blockDim.x + threadIdx.x;
    size_t stride = (size_t)gridDim.x * blockDim.x;
    for (; i < n; i += stride) out[i] = (__bf16)in[i];
}

// ---------------------------------------------------------------------------
// Embedding * sqrt(D) + sinusoidal positional encoding
// ---------------------------------------------------------------------------
__global__ void embed_kernel(const int* __restrict__ tgt, const float* __restrict__ table,
                             float* __restrict__ out) {
    int bt = blockIdx.x;           // b*T + t
    int t  = bt & (TT - 1);
    int tok = tgt[bt];
    const float* row = table + (size_t)tok * DM;
    const float sq = 27.712812921102035f;             // sqrt(768)
    const float nl = -9.210340371976184f / (float)DM; // -ln(10000)/D
    for (int d = threadIdx.x; d < DM; d += blockDim.x) {
        int i2 = (d >> 1) * 2;
        float div = __expf((float)i2 * nl);
        float ang = (float)t * div;
        float pe = (d & 1) ? __cosf(ang) : __sinf(ang);
        out[(size_t)bt * DM + d] = row[d] * sq + pe;
    }
}

// ---------------------------------------------------------------------------
// LayerNorm over last dim (D=768), one row per block. OutT = __bf16 or float.
// ---------------------------------------------------------------------------
template <typename OutT>
__global__ __launch_bounds__(256)
void ln_kernel(const float* __restrict__ x, const float* __restrict__ g,
               const float* __restrict__ beta, OutT* __restrict__ y) {
    int row = blockIdx.x;
    const float* xr = x + (size_t)row * DM;
    float s = 0.f, s2 = 0.f;
    for (int d = threadIdx.x; d < DM; d += 256) { float v = xr[d]; s += v; s2 += v * v; }
    __shared__ float red0[256];
    __shared__ float red1[256];
    red0[threadIdx.x] = s; red1[threadIdx.x] = s2;
    __syncthreads();
    for (int off = 128; off > 0; off >>= 1) {
        if (threadIdx.x < off) {
            red0[threadIdx.x] += red0[threadIdx.x + off];
            red1[threadIdx.x] += red1[threadIdx.x + off];
        }
        __syncthreads();
    }
    float mean = red0[0] * (1.0f / DM);
    float var  = red1[0] * (1.0f / DM) - mean * mean;
    float rs   = rsqrtf(var + 1e-6f);
    for (int d = threadIdx.x; d < DM; d += 256) {
        float v = (xr[d] - mean) * rs * g[d] + beta[d];
        y[(size_t)row * DM + d] = (OutT)v;
    }
}

// ---------------------------------------------------------------------------
// WMMA GEMM: y[M,N] = A[M,K](bf16) . W[N,K]^T(bf16) + bias[N], fused epilogue.
//   EPI 0: outF = v + resid            (f32, residual stream)
//   EPI 1: outB = gelu(v)              (bf16, FFN hidden)
//   EPI 2: outB[b,h,t,dh] = v          (bf16, head-split Q/K)
//   EPI 3: outB[b,h,dh,t] = v          (bf16, head-split transposed V)
// Block: 256 thr = 8 waves; tile 128x64; BK=32; wave tile 32x32 (4 WMMAs/step)
// A tile: TDM (tensor_load_to_lds) double-buffered, tracked by TENSORcnt.
// B tile: register-staged into LDS as [n][k] (stride 32) so frags load as 2x b128.
// ---------------------------------------------------------------------------
#define BTS 72   // sizes the Bt array only (32*BTS elems); B tile rows use stride 32

template <int EPI>
__global__ __launch_bounds__(256)
void gemm_bf16(const __bf16* __restrict__ A, const __bf16* __restrict__ W,
               const float* __restrict__ bias, int N, int K,
               float* __restrict__ outF, __bf16* __restrict__ outB,
               const float* __restrict__ resid) {
    const int tid  = threadIdx.x;
    const int lane = tid & 31;
    const int wave = tid >> 5;
    const int wm = (wave & 3) * 32;
    const int wn = (wave >> 2) * 32;
    const int m0 = blockIdx.y * 128;
    const int n0 = blockIdx.x * 64;

    __shared__ __bf16 As[2][128 * 32];  // double-buffered A tile
    __shared__ __bf16 Bt[32 * BTS];     // [n][k] weight tile, row stride 32

    v8f acc[2][2] = {};
    const int ml    = lane & 15;
    const int kbase = (lane < 16) ? 0 : 8;   // A-layout lane-split of K

    const int arow = tid >> 1, acol = (tid & 1) * 16;   // A stage (fallback)
    const int brow = tid >> 2, bcol = (tid & 3) * 8;    // B stage: 16B/thread

    const __bf16* aPtr = A + (size_t)(m0 + arow) * K + acol;
    const __bf16* wPtr = W + (size_t)(n0 + brow) * K + bcol;

#if USE_TDM
    if (wave == 0)
        tdm_load_tile_bf16((unsigned)(size_t)(void*)&As[0][0], A + (size_t)m0 * K,
                           32, 128, (unsigned)K, (unsigned)K);
#else
    bf16x8 ra0 = *(const bf16x8*)(aPtr);
    bf16x8 ra1 = *(const bf16x8*)(aPtr + 8);
#endif
    bf16x8 rb = *(const bf16x8*)(wPtr);

    int buf = 0;
    for (int k0 = 0; k0 < K; k0 += 32, buf ^= 1) {
        // commit staged B registers to LDS (as [n][k], k contiguous)
#pragma unroll
        for (int j = 0; j < 8; ++j)
            Bt[brow * 32 + bcol + j] = rb[j];
#if USE_TDM
        if (wave == 0) {
            if (k0 + 32 < K) { // async-load next A tile into the other buffer
                tdm_load_tile_bf16((unsigned)(size_t)(void*)&As[buf ^ 1][0],
                                   A + (size_t)m0 * K + k0 + 32,
                                   32, 128, (unsigned)K, (unsigned)K);
                // TENSORcnt completes in order: <=1 outstanding => current buf ready
                __builtin_amdgcn_s_wait_tensorcnt(1);
            } else {
                __builtin_amdgcn_s_wait_tensorcnt(0);
            }
        }
#else
        {
            bf16x8* da = (bf16x8*)(As[buf] + arow * 32 + acol);
            da[0] = ra0; da[1] = ra1;
        }
#endif
        __syncthreads();
        // issue next chunk's global loads while this chunk computes
        if (k0 + 32 < K) {
#if !USE_TDM
            ra0 = *(const bf16x8*)(aPtr + k0 + 32);
            ra1 = *(const bf16x8*)(aPtr + k0 + 40);
#endif
            rb = *(const bf16x8*)(wPtr + k0 + 32);
            if (k0 + 64 < K)   // hint chunk after next -> global_prefetch_b8
                __builtin_prefetch(wPtr + k0 + 64, 0, 1);
        }
        // A fragments (16x32, documented split-lane K layout) -> 2x b128 each
        v16bf af[2];
#pragma unroll
        for (int mt = 0; mt < 2; ++mt) {
            const __bf16* p = As[buf] + (wm + mt * 16 + ml) * 32 + kbase;
            union { v16bf v; bf16x8 h[2]; } u;
            u.h[0] = *(const bf16x8*)p;
            u.h[1] = *(const bf16x8*)(p + 16);
            af[mt] = u.v;
        }
        // B fragments (32x16: lane = N-col, elements = split-lane K) -> 2x b128 each
        v16bf bfrag[2];
#pragma unroll
        for (int nt = 0; nt < 2; ++nt) {
            const __bf16* pb = Bt + (wn + nt * 16 + ml) * 32 + kbase;
            union { v16bf v; bf16x8 h[2]; } u;
            u.h[0] = *(const bf16x8*)pb;
            u.h[1] = *(const bf16x8*)(pb + 16);
            bfrag[nt] = u.v;
        }
#pragma unroll
        for (int mt = 0; mt < 2; ++mt)
#pragma unroll
            for (int nt = 0; nt < 2; ++nt)
                acc[mt][nt] = __builtin_amdgcn_wmma_f32_16x16x32_bf16(
                    false, af[mt], false, bfrag[nt], (short)0, acc[mt][nt], false, false);
        __syncthreads();
    }

    // epilogue (C layout: VGPR r -> row r / r+8 by lane half; col = lane&15)
    const int rowHalf = (lane < 16) ? 0 : 8;
#pragma unroll
    for (int mt = 0; mt < 2; ++mt) {
        int rowBase = m0 + wm + mt * 16 + rowHalf;
#pragma unroll
        for (int nt = 0; nt < 2; ++nt) {
            int col = n0 + wn + nt * 16 + ml;
            float bia = bias[col];
#pragma unroll
            for (int r = 0; r < 8; ++r) {
                int mrow = rowBase + r;
                float v = acc[mt][nt][r] + bia;
                if (EPI == 0) {
                    size_t idx = (size_t)mrow * N + col;
                    outF[idx] = v + resid[idx];
                } else if (EPI == 1) {
                    outB[(size_t)mrow * N + col] = (__bf16)gelu_f(v);
                } else if (EPI == 2) {
                    int b = mrow >> 9, t = mrow & (TT - 1);   // T = 512
                    int h = col >> 6,  d = col & 63;          // DH = 64
                    outB[(((size_t)b * NH + h) * TT + t) * DH + d] = (__bf16)v;
                } else {
                    int b = mrow >> 9, t = mrow & (TT - 1);
                    int h = col >> 6,  d = col & 63;
                    outB[(((size_t)b * NH + h) * DH + d) * TT + t] = (__bf16)v;
                }
            }
        }
    }
}

// ---------------------------------------------------------------------------
// Fused attention: one wave handles 16 query rows of one (b,h).
// Q,K: bf16 [B,H,Tkv,64]; V: bf16 [B,H,64,Tkv] (pre-transposed by GEMM EPI=3).
// Online softmax: row-max via v_permlane16 butterflies (VALU), softmax
// denominator accumulated by an extra ones-column WMMA (same rescale
// recurrence as ctx), causal tiles beyond the query tile skipped entirely.
// Operand loads batched ahead of WMMA chains so they clause + overlap.
// ---------------------------------------------------------------------------
__global__ __launch_bounds__(32)
void attn_kernel(const __bf16* __restrict__ Q, const __bf16* __restrict__ Kt,
                 const __bf16* __restrict__ Vh, const int* __restrict__ toks,
                 __bf16* __restrict__ ctx, int Tkv, int causal, float scale) {
    const int lane  = threadIdx.x & 31;
    const int qrow0 = blockIdx.x * 16;
    const int bh = blockIdx.y;
    const int b = bh / NH, h = bh % NH;
    const int ml = lane & 15;
    const int half8 = (lane < 16) ? 0 : 8;
    const int kbase = half8;   // A-layout K split (0 or 8)

    // Q fragments: A 16x32 for dh-chunks 0 and 1 (loaded once)
    const __bf16* Qbase = Q + (((size_t)b * NH + h) * TT + qrow0) * DH;
    v16bf aQ[2];
#pragma unroll
    for (int c = 0; c < 2; ++c) {
        const __bf16* qr = Qbase + (size_t)ml * DH + c * 32 + kbase;
        union { v16bf v; bf16x8 hh[2]; } u;
        u.hh[0] = *(const bf16x8*)qr;
        u.hh[1] = *(const bf16x8*)(qr + 16);
        aQ[c] = u.v;
    }

    v16bf ones;
#pragma unroll
    for (int e = 0; e < 16; ++e) ones[e] = (__bf16)1.0f;

    __shared__ __bf16 Ps[16 * 32];

    v8f acc[4] = {};
    v8f accL = {};            // ones-column accumulator: ends as softmax denom
    float mrow[8];
#pragma unroll
    for (int r = 0; r < 8; ++r) mrow[r] = -__builtin_inff();

    const __bf16* Ktbase = Kt + ((size_t)b * NH + h) * DH * Tkv; // [key][dh]
    const __bf16* Vbase  = Vh + ((size_t)b * NH + h) * (size_t)Tkv * DH;

    // causal: keys > qrow0+15 are masked for the entire tile -> skip them
    const int kend = causal ? (qrow0 + 16) : Tkv;

    for (int ks = 0; ks < kend; ks += 32) {
        // batch all 4 K fragments (B 32x16: lane = key col, elems = 16 dh)
        union U { v16bf v; bf16x8 hh[2]; };
        U kf[4];
#pragma unroll
        for (int j = 0; j < 4; ++j) {
            int hlf = j >> 1, c = j & 1;
            const __bf16* kr = Ktbase + (size_t)(ks + hlf * 16 + ml) * DH + c * 32 + kbase;
            kf[j].hh[0] = *(const bf16x8*)kr;
            kf[j].hh[1] = *(const bf16x8*)(kr + 16);
        }
        v8f sc[2] = {};
#pragma unroll
        for (int j = 0; j < 4; ++j) {
            int hlf = j >> 1, c = j & 1;
            sc[hlf] = __builtin_amdgcn_wmma_f32_16x16x32_bf16(
                false, aQ[c], false, kf[j].v, (short)0, sc[hlf], false, false);
        }

        // mask + online softmax (rows r / r+8 by lane half; cols across 16 lanes)
        int key0 = ks + ml, key1 = ks + 16 + ml;
        int tok0 = toks[b * Tkv + key0];
        int tok1 = toks[b * Tkv + key1];
#pragma unroll
        for (int r = 0; r < 8; ++r) {
            int qi = qrow0 + half8 + r;
            bool m0b = (tok0 == 0) || (causal && (key0 > qi));
            bool m1b = (tok1 == 0) || (causal && (key1 > qi));
            float v0 = m0b ? -1e18f : sc[0][r] * scale;
            float v1 = m1b ? -1e18f : sc[1][r] * scale;
            float mnew = fmaxf(mrow[r], rowmax16(fmaxf(v0, v1)));
            float f  = __expf(mrow[r] - mnew);
            mrow[r] = mnew;
            Ps[(half8 + r) * 32 + ml]      = (__bf16)__expf(v0 - mnew);
            Ps[(half8 + r) * 32 + 16 + ml] = (__bf16)__expf(v1 - mnew);
#pragma unroll
            for (int nt = 0; nt < 4; ++nt) acc[nt][r] *= f;
            accL[r] *= f;
        }
        __syncthreads();
        // P (16x32) A-layout fragment (2x ds_load_b128)
        v16bf pA;
        {
            const __bf16* pr = Ps + ml * 32 + kbase;
            union { v16bf v; bf16x8 hh[2]; } u;
            u.hh[0] = *(const bf16x8*)pr;
            u.hh[1] = *(const bf16x8*)(pr + 16);
            pA = u.v;
        }

        // batch V fragments, then ctx(16x64) += P(16x32) . V(32x64)
        U vf[4];
#pragma unroll
        for (int nt = 0; nt < 4; ++nt) {
            const __bf16* vr = Vbase + (size_t)(nt * 16 + ml) * Tkv + ks + kbase;
            vf[nt].hh[0] = *(const bf16x8*)vr;
            vf[nt].hh[1] = *(const bf16x8*)(vr + 16);
        }
#pragma unroll
        for (int nt = 0; nt < 4; ++nt)
            acc[nt] = __builtin_amdgcn_wmma_f32_16x16x32_bf16(
                false, pA, false, vf[nt].v, (short)0, acc[nt], false, false);
        accL = __builtin_amdgcn_wmma_f32_16x16x32_bf16(
            false, pA, false, ones, (short)0, accL, false, false);
        __syncthreads();
    }

    // normalize & write ctx as bf16 [B,T,D] at column h*64
#pragma unroll
    for (int r = 0; r < 8; ++r) {
        float inv = 1.0f / accL[r];
        int row = qrow0 + half8 + r;
        size_t base = ((size_t)b * TT + row) * DM + h * DH;
#pragma unroll
        for (int nt = 0; nt < 4; ++nt)
            ctx[base + nt * 16 + ml] = (__bf16)(acc[nt][r] * inv);
    }
}

// ---------------------------------------------------------------------------
// Host orchestration
// ---------------------------------------------------------------------------
extern "C" void kernel_launch(void* const* d_in, const int* in_sizes, int n_in,
                              void* d_out, int out_size, void* d_ws, size_t ws_size,
                              hipStream_t stream) {
    (void)in_sizes; (void)n_in; (void)out_size; (void)ws_size;
    const int*   tgt   = (const int*)d_in[0];
    const int*   srct  = (const int*)d_in[1];
    const float* memb  = (const float*)d_in[2];
    const float* etab  = (const float*)d_in[3];
    const float* sa_w  = (const float*)d_in[4];
    const float* sa_b  = (const float*)d_in[5];
    const float* ca_w  = (const float*)d_in[6];
    const float* ca_b  = (const float*)d_in[7];
    const float* ln_g  = (const float*)d_in[8];
    const float* ln_bb = (const float*)d_in[9];
    const float* ff_w1 = (const float*)d_in[10];
    const float* ff_b1 = (const float*)d_in[11];
    const float* ff_w2 = (const float*)d_in[12];
    const float* ff_b2 = (const float*)d_in[13];
    const float* out_g = (const float*)d_in[14];
    const float* out_b = (const float*)d_in[15];

    // workspace carve
    char* p = (char*)d_ws;
    auto carve = [&](size_t bytes) -> char* {
        char* r = p; p += (bytes + 255) & ~(size_t)255; return r;
    };
    const size_t nSAW = (size_t)LAY * 4 * DM * DM;
    const size_t nFF1 = (size_t)LAY * DFF * DM;
    __bf16* saWb  = (__bf16*)carve(nSAW * 2);
    __bf16* caWb  = (__bf16*)carve(nSAW * 2);
    __bf16* w1b   = (__bf16*)carve(nFF1 * 2);
    __bf16* w2b   = (__bf16*)carve(nFF1 * 2);
    __bf16* membB = (__bf16*)carve((size_t)BB * SS * DM * 2);
    float*  bufA  = (float*)carve((size_t)MM * DM * 4);
    float*  bufB  = (float*)carve((size_t)MM * DM * 4);
    __bf16* Xb    = (__bf16*)carve((size_t)MM * DM * 2);
    __bf16* Qp    = (__bf16*)carve((size_t)MM * DM * 2);
    __bf16* Kp    = (__bf16*)carve((size_t)MM * DM * 2);  // [B,H,T,DH]
    __bf16* Vp    = (__bf16*)carve((size_t)MM * DM * 2);
    __bf16* Ctx   = (__bf16*)carve((size_t)MM * DM * 2);
    __bf16* Hb    = (__bf16*)carve((size_t)MM * DFF * 2);

    // one-time (per launch) bf16 conversions
    cvt_f32_bf16<<<2048, 256, 0, stream>>>(sa_w,  saWb,  nSAW);
    cvt_f32_bf16<<<2048, 256, 0, stream>>>(ca_w,  caWb,  nSAW);
    cvt_f32_bf16<<<2048, 256, 0, stream>>>(ff_w1, w1b,   nFF1);
    cvt_f32_bf16<<<2048, 256, 0, stream>>>(ff_w2, w2b,   nFF1);
    cvt_f32_bf16<<<2048, 256, 0, stream>>>(memb,  membB, (size_t)BB * SS * DM);

    float* cur = bufA;
    float* alt = bufB;
    embed_kernel<<<MM, 256, 0, stream>>>(tgt, etab, cur);

    const dim3 gD(DM / 64, MM / 128);     // N=768 GEMMs
    const dim3 gF(DFF / 64, MM / 128);    // N=3072 GEMM
    const dim3 gAttn(TT / 16, BB * NH);
    const float scale = 0.125f;           // DH^-0.5

    for (int i = 0; i < LAY; ++i) {
        const __bf16* sw = saWb + (size_t)i * 4 * DM * DM;
        const float*  sb = sa_b + (size_t)i * 4 * DM;
        const __bf16* cw = caWb + (size_t)i * 4 * DM * DM;
        const float*  cb = ca_b + (size_t)i * 4 * DM;

        // --- self attention (pre-LN) ---
        ln_kernel<__bf16><<<MM, 256, 0, stream>>>(cur, ln_g + (i * 3 + 0) * DM,
                                                  ln_bb + (i * 3 + 0) * DM, Xb);
        gemm_bf16<2><<<gD, 256, 0, stream>>>(Xb, sw + 0 * DM * DM, sb + 0 * DM, DM, DM,
                                             nullptr, Qp, nullptr);
        gemm_bf16<2><<<gD, 256, 0, stream>>>(Xb, sw + 1 * DM * DM, sb + 1 * DM, DM, DM,
                                             nullptr, Kp, nullptr);
        gemm_bf16<3><<<gD, 256, 0, stream>>>(Xb, sw + 2 * DM * DM, sb + 2 * DM, DM, DM,
                                             nullptr, Vp, nullptr);
        attn_kernel<<<gAttn, 32, 0, stream>>>(Qp, Kp, Vp, tgt, Ctx, TT, 1, scale);
        gemm_bf16<0><<<gD, 256, 0, stream>>>(Ctx, sw + 3 * DM * DM, sb + 3 * DM, DM, DM,
                                             alt, nullptr, cur);
        { float* t = cur; cur = alt; alt = t; }

        // --- cross attention ---
        ln_kernel<__bf16><<<MM, 256, 0, stream>>>(cur, ln_g + (i * 3 + 1) * DM,
                                                  ln_bb + (i * 3 + 1) * DM, Xb);
        gemm_bf16<2><<<gD, 256, 0, stream>>>(Xb,    cw + 0 * DM * DM, cb + 0 * DM, DM, DM,
                                             nullptr, Qp, nullptr);
        gemm_bf16<2><<<gD, 256, 0, stream>>>(membB, cw + 1 * DM * DM, cb + 1 * DM, DM, DM,
                                             nullptr, Kp, nullptr);
        gemm_bf16<3><<<gD, 256, 0, stream>>>(membB, cw + 2 * DM * DM, cb + 2 * DM, DM, DM,
                                             nullptr, Vp, nullptr);
        attn_kernel<<<gAttn, 32, 0, stream>>>(Qp, Kp, Vp, srct, Ctx, SS, 0, scale);
        gemm_bf16<0><<<gD, 256, 0, stream>>>(Ctx, cw + 3 * DM * DM, cb + 3 * DM, DM, DM,
                                             alt, nullptr, cur);
        { float* t = cur; cur = alt; alt = t; }

        // --- FFN ---
        ln_kernel<__bf16><<<MM, 256, 0, stream>>>(cur, ln_g + (i * 3 + 2) * DM,
                                                  ln_bb + (i * 3 + 2) * DM, Xb);
        gemm_bf16<1><<<gF, 256, 0, stream>>>(Xb, w1b + (size_t)i * DFF * DM,
                                             ff_b1 + (size_t)i * DFF, DFF, DM,
                                             nullptr, Hb, nullptr);
        gemm_bf16<0><<<gD, 256, 0, stream>>>(Hb, w2b + (size_t)i * DM * DFF,
                                             ff_b2 + (size_t)i * DM, DM, DFF,
                                             alt, nullptr, cur);
        { float* t = cur; cur = alt; alt = t; }
    }

    ln_kernel<float><<<MM, 256, 0, stream>>>(cur, out_g, out_b, (float*)d_out);
}
